// HyperSSMLayer_62302795596488
// MI455X (gfx1250) — compile-verified, matrix operand order checked by reading, not run on hardware
//
#include <hip/hip_runtime.h>
#include <hip/hip_bf16.h>
#include <math.h>

#define BB 8
#define SS 2048
#define DD 1024
#define HH 4
#define NN 8
#define ROWS (BB*SS)              /* 16384 tokens */
#define HN (HH*NN)                /* 32  */
#define HNN (HH*NN*NN)            /* 256 */
#define PC (HN + HNN + HN + HN)   /* 352 projection cols: sign|perm|gate|in */
#define LN_EPS 1e-5f

typedef __attribute__((ext_vector_type(16))) _Float16 v16h;
typedef __attribute__((ext_vector_type(8)))  _Float16 v8h;
typedef __attribute__((ext_vector_type(8)))  float    v8f;

// ---------------------------------------------------------------------------
// Kernel 1: LayerNorm (per token row of 1024) -> f16
// ---------------------------------------------------------------------------
__global__ __launch_bounds__(256)
void hyperssm_ln(const float* __restrict__ x,
                 const float* __restrict__ ln_w,
                 const float* __restrict__ ln_b,
                 _Float16* __restrict__ xn) {
  const int row = blockIdx.x;
  const float* xr = x + (size_t)row * DD;
  float4 v = ((const float4*)xr)[threadIdx.x];           // 256 thr * 4 = 1024
  float s  = v.x + v.y + v.z + v.w;
  float s2 = v.x*v.x + v.y*v.y + v.z*v.z + v.w*v.w;
  __shared__ float red[2][8];
  #pragma unroll
  for (int off = 16; off > 0; off >>= 1) {
    s  += __shfl_down(s,  off, 32);
    s2 += __shfl_down(s2, off, 32);
  }
  const int wid = threadIdx.x >> 5, lane = threadIdx.x & 31;
  if (lane == 0) { red[0][wid] = s; red[1][wid] = s2; }
  __syncthreads();
  if (wid == 0) {
    s  = (lane < 8) ? red[0][lane] : 0.f;
    s2 = (lane < 8) ? red[1][lane] : 0.f;
    #pragma unroll
    for (int off = 4; off > 0; off >>= 1) {
      s  += __shfl_down(s,  off, 32);
      s2 += __shfl_down(s2, off, 32);
    }
    if (lane == 0) { red[0][0] = s; red[1][0] = s2; }
  }
  __syncthreads();
  const float mu   = red[0][0] * (1.0f/DD);
  const float var  = red[1][0] * (1.0f/DD) - mu*mu;
  const float rstd = rsqrtf(var + LN_EPS);
  const int c0 = threadIdx.x * 4;
  _Float16* o = xn + (size_t)row*DD + c0;
  o[0] = (_Float16)(((v.x - mu)*rstd) * ln_w[c0+0] + ln_b[c0+0]);
  o[1] = (_Float16)(((v.y - mu)*rstd) * ln_w[c0+1] + ln_b[c0+1]);
  o[2] = (_Float16)(((v.z - mu)*rstd) * ln_w[c0+2] + ln_b[c0+2]);
  o[3] = (_Float16)(((v.w - mu)*rstd) * ln_w[c0+3] + ln_b[c0+3]);
}

// ---------------------------------------------------------------------------
// Kernel 2: weight prep. wtp[n][k] = W_cat[k][n] (f16, 352x1024),
//           wto[n][k] = out_W[k][n] (f16, 1024x32), pbias[352].
// ---------------------------------------------------------------------------
__global__ __launch_bounds__(256)
void hyperssm_prep(const float* __restrict__ sign_W, const float* __restrict__ sign_b,
                   const float* __restrict__ perm_W, const float* __restrict__ perm_b,
                   const float* __restrict__ gate_W, const float* __restrict__ gate_b,
                   const float* __restrict__ in_W,   const float* __restrict__ in_b,
                   const float* __restrict__ out_W,
                   _Float16* __restrict__ wtp, _Float16* __restrict__ wto,
                   float* __restrict__ pbias) {
  const int row = blockIdx.x;
  if (row < PC) {
    const float* src; int stride; float bias;
    if (row < HN)              { src = sign_W + row;            stride = HN;  bias = sign_b[row]; }
    else if (row < HN+HNN)     { int m = row-HN;       src = perm_W + m; stride = HNN; bias = perm_b[m]; }
    else if (row < HN+HNN+HN)  { int m = row-HN-HNN;   src = gate_W + m; stride = HN;  bias = gate_b[m]; }
    else                       { int m = row-HN-HNN-HN;src = in_W   + m; stride = HN;  bias = in_b[m]; }
    for (int k = threadIdx.x; k < DD; k += blockDim.x)
      wtp[(size_t)row*DD + k] = (_Float16)src[(size_t)k*stride];
    if (threadIdx.x == 0) pbias[row] = bias;
  } else {
    const int n = row - PC;                       // output column 0..1023
    for (int k = threadIdx.x; k < HN; k += blockDim.x)
      wto[(size_t)n*HN + k] = (_Float16)out_W[(size_t)k*DD + n];
  }
}

// ---------------------------------------------------------------------------
// WMMA fragment load: 16-bit A/B fragment, row-major-K source.
// lane&15 selects row (M or N), lane>>4 selects K half (0/8), +16 upper quarter.
// ---------------------------------------------------------------------------
__device__ __forceinline__ v16h load_frag16(const _Float16* __restrict__ rowp, int koff) {
  const v8h lo = *(const v8h*)(rowp + koff);
  const v8h hi = *(const v8h*)(rowp + koff + 16);
  v16h r;
  #pragma unroll
  for (int j = 0; j < 8; ++j) { r[j] = lo[j]; r[8+j] = hi[j]; }
  return r;
}

// ---------------------------------------------------------------------------
// Kernel 3: projection GEMM. M=16384, N=352, K=1024.
// 2x2 register blocking: one wave computes a 32x32 output block
// (4 accumulators, 4 WMMAs per K-step at 2 b128 loads per WMMA).
// proj = xn @ W_cat + bias   (f32 result)
// ---------------------------------------------------------------------------
__global__ __launch_bounds__(256)
void hyperssm_gemm_proj(const _Float16* __restrict__ xn,
                        const _Float16* __restrict__ wtp,
                        const float* __restrict__ pbias,
                        float* __restrict__ proj) {
  const int wg    = blockIdx.x * (blockDim.x >> 5) + (threadIdx.x >> 5);
  const int lane  = threadIdx.x & 31;
  const int pairM = wg / (PC/32);                 // 0..511 (PC/32 = 11)
  const int pairN = wg % (PC/32);                 // 0..10
  const int half  = lane >> 4;
  const int mr    = lane & 15;
  const _Float16* arow0 = xn  + (size_t)(pairM*32 + mr)      * DD;
  const _Float16* arow1 = xn  + (size_t)(pairM*32 + 16 + mr) * DD;
  const _Float16* brow0 = wtp + (size_t)(pairN*32 + mr)      * DD;
  const _Float16* brow1 = wtp + (size_t)(pairN*32 + 16 + mr) * DD;
  v8f acc00 = {}, acc01 = {}, acc10 = {}, acc11 = {};
  for (int k0 = 0; k0 < DD; k0 += 32) {
    __builtin_prefetch(arow0 + k0 + 128, 0, 1);    // global_prefetch_b8
    __builtin_prefetch(arow1 + k0 + 128, 0, 1);
    const int off = k0 + half*8;
    v16h a0 = load_frag16(arow0, off);
    v16h a1 = load_frag16(arow1, off);
    v16h b0 = load_frag16(brow0, off);
    v16h b1 = load_frag16(brow1, off);
    acc00 = __builtin_amdgcn_wmma_f32_16x16x32_f16(false, a0, false, b0,
                                                   (short)0, acc00, false, false);
    acc01 = __builtin_amdgcn_wmma_f32_16x16x32_f16(false, a0, false, b1,
                                                   (short)0, acc01, false, false);
    acc10 = __builtin_amdgcn_wmma_f32_16x16x32_f16(false, a1, false, b0,
                                                   (short)0, acc10, false, false);
    acc11 = __builtin_amdgcn_wmma_f32_16x16x32_f16(false, a1, false, b1,
                                                   (short)0, acc11, false, false);
  }
  const int col0 = pairN*32 + mr;
  const int col1 = col0 + 16;
  const float bias0 = pbias[col0];
  const float bias1 = pbias[col1];
  #pragma unroll
  for (int r = 0; r < 8; ++r) {
    const int m0 = pairM*32 + r + half*8;          // C layout: VGPR r -> M=r / r+8
    const int m1 = m0 + 16;
    proj[(size_t)m0*PC + col0] = acc00[r] + bias0;
    proj[(size_t)m0*PC + col1] = acc01[r] + bias1;
    proj[(size_t)m1*PC + col0] = acc10[r] + bias0;
    proj[(size_t)m1*PC + col1] = acc11[r] + bias1;
  }
}

// ---------------------------------------------------------------------------
// Kernel 4: Sinkhorn (log space, 5 iters) + greedy hard assignment.
// One thread per (token,head) 8x8 matrix -> 8 packed permutation indices.
// Forward value of P = hard (straight-through), so indices suffice.
// ---------------------------------------------------------------------------
__global__ __launch_bounds__(256)
void hyperssm_sinkhorn(const float* __restrict__ proj,
                       unsigned long long* __restrict__ pidx) {
  const int t = blockIdx.x * blockDim.x + threadIdx.x;   // row*H + h
  if (t >= ROWS*HH) return;
  const int row = t >> 2;
  const int h   = t & 3;
  const float* base = proj + (size_t)row*PC + HN + h*(NN*NN);  // TAU = 1
  float a[NN][NN];
  #pragma unroll
  for (int i = 0; i < NN; ++i)
    #pragma unroll
    for (int j = 0; j < NN; ++j) a[i][j] = base[i*NN + j];
  #pragma unroll
  for (int it = 0; it < 5; ++it) {
    #pragma unroll
    for (int i = 0; i < NN; ++i) {                 // row logsumexp
      float m = a[i][0];
      #pragma unroll
      for (int j = 1; j < NN; ++j) m = fmaxf(m, a[i][j]);
      float s = 0.f;
      #pragma unroll
      for (int j = 0; j < NN; ++j) s += __expf(a[i][j] - m);
      const float lse = m + __logf(s);
      #pragma unroll
      for (int j = 0; j < NN; ++j) a[i][j] -= lse;
    }
    #pragma unroll
    for (int j = 0; j < NN; ++j) {                 // col logsumexp
      float m = a[0][j];
      #pragma unroll
      for (int i = 1; i < NN; ++i) m = fmaxf(m, a[i][j]);
      float s = 0.f;
      #pragma unroll
      for (int i = 0; i < NN; ++i) s += __expf(a[i][j] - m);
      const float lse = m + __logf(s);
      #pragma unroll
      for (int i = 0; i < NN; ++i) a[i][j] -= lse;
    }
  }
  // greedy: rows in order of descending row-max (exp is monotone -> use logs)
  float rmax[NN];
  #pragma unroll
  for (int i = 0; i < NN; ++i) {
    float m = a[i][0];
    #pragma unroll
    for (int j = 1; j < NN; ++j) m = fmaxf(m, a[i][j]);
    rmax[i] = m;
  }
  unsigned rowdone = 0, avail = 0xFFu;
  unsigned long long packed = 0;
  for (int step = 0; step < NN; ++step) {
    float best = -INFINITY; int r = 0;
    for (int i = 0; i < NN; ++i)
      if (!((rowdone >> i) & 1u) && rmax[i] > best) { best = rmax[i]; r = i; }
    float bc = -INFINITY; int c = 0;
    for (int j = 0; j < NN; ++j)
      if (((avail >> j) & 1u) && a[r][j] > bc) { bc = a[r][j]; c = j; }
    avail &= ~(1u << c);
    rowdone |= 1u << r;
    packed |= ((unsigned long long)c) << (8*r);
  }
  pidx[t] = packed;
}

// ---------------------------------------------------------------------------
// Kernel 5: sequential scan. 32 chains (B*H), 4 chains/wave, 8 waves, 1 block.
// h_new[i] = sigmoid(gz)*(2*sigmoid(sz)-1)*h[p[i]] + (1-sigmoid(gz))*b_in
// Critical path per step = shuffle + few VALU; loads are state-independent.
// ---------------------------------------------------------------------------
__global__ __launch_bounds__(256)
void hyperssm_scan(const float* __restrict__ proj,
                   const unsigned long long* __restrict__ pidx,
                   _Float16* __restrict__ ys) {
  const int lane  = threadIdx.x & 31;
  const int wid   = threadIdx.x >> 5;
  const int chain = wid*4 + (lane >> 3);   // 0..31 = b*H + h
  const int i     = lane & 7;
  const int b     = chain >> 2;
  const int h     = chain & 3;
  const int colS  = h*NN + i;
  const int colG  = HN + HNN + h*NN + i;
  const int colB  = HN + HNN + HN + h*NN + i;
  const int laneBase = lane & ~7;
  float hv = 0.f;
  for (int t = 0; t < SS; ++t) {
    const size_t row = (size_t)b*SS + t;
    const float* pr = proj + row*PC;
    const float sz = pr[colS];
    const float gz = pr[colG];
    const float bz = pr[colB];
    const unsigned long long pk = pidx[row*HH + h];
    const int p  = (int)((pk >> (8*i)) & 7ull);
    const float hp = __shfl(hv, laneBase + p, 32);
    const float s  = 2.f / (1.f + __expf(-sz)) - 1.f;   // 2*sigmoid-1
    const float g  = 1.f / (1.f + __expf(-gz));
    hv = g*(s*hp) + (1.f - g)*bz;
    ys[row*HN + h*NN + i] = (_Float16)hv;
  }
}

// ---------------------------------------------------------------------------
// Kernel 6: output GEMM fused with bias + residual.
// M=16384, N=1024, K=32. 2x2 blocking: 4 WMMAs per wave (single K step),
// fragment loads amortized across the 2x2 tile block.
// ---------------------------------------------------------------------------
__global__ __launch_bounds__(256)
void hyperssm_gemm_out(const _Float16* __restrict__ ys,
                       const _Float16* __restrict__ wto,
                       const float* __restrict__ out_b,
                       const float* __restrict__ x,
                       float* __restrict__ out) {
  const int wg    = blockIdx.x * (blockDim.x >> 5) + (threadIdx.x >> 5);
  const int lane  = threadIdx.x & 31;
  const int pairM = wg / (DD/32);                  // 0..511 (DD/32 = 32)
  const int pairN = wg % (DD/32);                  // 0..31
  const int half  = lane >> 4;
  const int mr    = lane & 15;
  const int koff  = half*8;
  v16h a0 = load_frag16(ys  + (size_t)(pairM*32 + mr)      * HN, koff);
  v16h a1 = load_frag16(ys  + (size_t)(pairM*32 + 16 + mr) * HN, koff);
  v16h b0 = load_frag16(wto + (size_t)(pairN*32 + mr)      * HN, koff);
  v16h b1 = load_frag16(wto + (size_t)(pairN*32 + 16 + mr) * HN, koff);
  v8f acc00 = {}, acc01 = {}, acc10 = {}, acc11 = {};
  acc00 = __builtin_amdgcn_wmma_f32_16x16x32_f16(false, a0, false, b0,
                                                 (short)0, acc00, false, false);
  acc01 = __builtin_amdgcn_wmma_f32_16x16x32_f16(false, a0, false, b1,
                                                 (short)0, acc01, false, false);
  acc10 = __builtin_amdgcn_wmma_f32_16x16x32_f16(false, a1, false, b0,
                                                 (short)0, acc10, false, false);
  acc11 = __builtin_amdgcn_wmma_f32_16x16x32_f16(false, a1, false, b1,
                                                 (short)0, acc11, false, false);
  const int col0 = pairN*32 + mr;
  const int col1 = col0 + 16;
  const float bias0 = out_b[col0];
  const float bias1 = out_b[col1];
  #pragma unroll
  for (int r = 0; r < 8; ++r) {
    const size_t m0 = (size_t)pairM*32 + r + half*8;
    const size_t m1 = m0 + 16;
    const size_t i00 = m0*DD + col0, i01 = m0*DD + col1;
    const size_t i10 = m1*DD + col0, i11 = m1*DD + col1;
    out[i00] = acc00[r] + bias0 + x[i00];
    out[i01] = acc01[r] + bias1 + x[i01];
    out[i10] = acc10[r] + bias0 + x[i10];
    out[i11] = acc11[r] + bias1 + x[i11];
  }
}

// ---------------------------------------------------------------------------
extern "C" void kernel_launch(void* const* d_in, const int* in_sizes, int n_in,
                              void* d_out, int out_size, void* d_ws, size_t ws_size,
                              hipStream_t stream) {
  const float* x      = (const float*)d_in[0];
  const float* ln_w   = (const float*)d_in[1];
  const float* ln_b   = (const float*)d_in[2];
  const float* sign_W = (const float*)d_in[3];
  const float* sign_b = (const float*)d_in[4];
  const float* perm_W = (const float*)d_in[5];
  const float* perm_b = (const float*)d_in[6];
  const float* gate_W = (const float*)d_in[7];
  const float* gate_b = (const float*)d_in[8];
  const float* in_W   = (const float*)d_in[9];
  const float* in_b   = (const float*)d_in[10];
  const float* out_W  = (const float*)d_in[11];
  const float* out_b  = (const float*)d_in[12];
  float* out = (float*)d_out;

  // workspace layout (~56.3 MB total)
  char* p = (char*)d_ws;
  _Float16* xn  = (_Float16*)p;           p += (size_t)ROWS*DD*sizeof(_Float16);
  _Float16* wtp = (_Float16*)p;           p += (size_t)PC*DD*sizeof(_Float16);
  _Float16* wto = (_Float16*)p;           p += (size_t)DD*HN*sizeof(_Float16);
  float* pbias  = (float*)p;              p += 2048;
  float* proj   = (float*)p;              p += (size_t)ROWS*PC*sizeof(float);
  unsigned long long* pidx = (unsigned long long*)p;
                                          p += (size_t)ROWS*HH*sizeof(unsigned long long);
  _Float16* ys  = (_Float16*)p;           p += (size_t)ROWS*HN*sizeof(_Float16);

  hyperssm_ln<<<ROWS, 256, 0, stream>>>(x, ln_w, ln_b, xn);
  hyperssm_prep<<<PC + DD, 256, 0, stream>>>(sign_W, sign_b, perm_W, perm_b,
                                             gate_W, gate_b, in_W, in_b,
                                             out_W, wtp, wto, pbias);
  // (ROWS/32)*(PC/32) = 512*11 = 5632 waves, 8 waves/block -> 704 blocks
  hyperssm_gemm_proj<<<(ROWS/32)*(PC/32)/8, 256, 0, stream>>>(xn, wtp, pbias, proj);
  hyperssm_sinkhorn<<<(ROWS*HH)/256, 256, 0, stream>>>(proj, pidx);
  hyperssm_scan<<<1, 256, 0, stream>>>(proj, pidx, ys);
  // (ROWS/32)*(DD/32) = 512*32 = 16384 waves -> 2048 blocks
  hyperssm_gemm_out<<<(ROWS/32)*(DD/32)/8, 256, 0, stream>>>(ys, wto, out_b, x, out);
}
